// Attention_3822520894100
// MI455X (gfx1250) — matrix operand order repaired and checked by reading, not compiled, not run on hardware
//
#include <hip/hip_runtime.h>

// ---------------------------------------------------------------------------
// MI455X (gfx1250) attention block: LN -> QKV GEMM -> exp-attn -> out GEMM
// All matmuls via v_wmma_f32_16x16x32_bf16 (wave32, f32 accumulate).
// Round 2: 32-row M tiles everywhere => every B fragment feeds 2 WMMAs
// (2x arithmetic intensity vs round 1).
// ---------------------------------------------------------------------------

typedef __attribute__((ext_vector_type(16))) __bf16 v16bf;
typedef __attribute__((ext_vector_type(8)))  __bf16 v8bf;
typedef __attribute__((ext_vector_type(8)))  float  v8f;
typedef __attribute__((ext_vector_type(4)))  float  v4f;

static constexpr int   Bc    = 2;
static constexpr int   Nc    = 2048;
static constexpr int   DIMc  = 1024;
static constexpr int   Hc    = 16;
static constexpr int   DHc   = 64;
static constexpr int   INNERc= 1024;     // H*DH
static constexpr float LN_EPS  = 1e-5f;
static constexpr float ATTN_EPS= 1e-8f;
static constexpr float SCALEc  = 0.125f; // 64^-0.5

__device__ __forceinline__ __bf16 to_bf16(float f) {
    unsigned u = __builtin_bit_cast(unsigned, f);
    unsigned r = (u + 0x7FFFu + ((u >> 16) & 1u)) >> 16;   // RNE
    unsigned short s = (unsigned short)r;
    return __builtin_bit_cast(__bf16, s);
}

// A-matrix fragment, 16(M) x 32(K), bf16, row-major source with leading dim ld.
// ISA 7.12.2: lane m=lane&15, g=lane>>4 holds K = {8g..8g+7} U {16+8g..16+8g+7}.
__device__ __forceinline__ v16bf load_a16x32(const __bf16* base, int ld) {
    int lane = threadIdx.x;
    int m = lane & 15, g = lane >> 4;
    const __bf16* row = base + (size_t)m * ld + 8 * g;
    union { v16bf v; v8bf h[2]; } u;
    u.h[0] = *(const v8bf*)(row);
    u.h[1] = *(const v8bf*)(row + 16);
    return u.v;
}

// B-matrix fragment, 32(K) x 16(N), from a TRANSPOSED (N-major) source:
// baseT[n][k], leading dim ld. Lane n=lane&15, g=lane>>4 holds K = {8g..8g+7} U {16+8g..16+8g+7}.
__device__ __forceinline__ v16bf load_b32x16T(const __bf16* baseT, int ld) {
    int lane = threadIdx.x;
    int n = lane & 15, g = lane >> 4;
    const __bf16* p = baseT + (size_t)n * ld + 8 * g;
    union { v16bf v; v8bf h[2]; } u;
    u.h[0] = *(const v8bf*)(p);
    u.h[1] = *(const v8bf*)(p + 16);
    return u.v;
}

__device__ __forceinline__ v8f wmma_bf16(v16bf a, v16bf b, v8f c) {
    return __builtin_amdgcn_wmma_f32_16x16x32_bf16(false, a, false, b,
                                                   (short)0, c, false, false);
}

// --------------------------- LayerNorm -> bf16 -----------------------------
__global__ __launch_bounds__(128) void ln_kernel(const float* __restrict__ x,
                                                 const float* __restrict__ gam,
                                                 const float* __restrict__ bet,
                                                 __bf16* __restrict__ xn) {
    int row  = blockIdx.x * blockDim.y + threadIdx.y;   // 0..4095
    int lane = threadIdx.x;
    const float* xr = x + (size_t)row * DIMc;
    float vals[32];
    float sum = 0.f, sq = 0.f;
#pragma unroll
    for (int i = 0; i < 8; ++i) {
        v4f v = *(const v4f*)(xr + lane * 4 + i * 128);
#pragma unroll
        for (int j = 0; j < 4; ++j) {
            float f = v[j];
            vals[i * 4 + j] = f;
            sum += f; sq += f * f;
        }
    }
#pragma unroll
    for (int m = 16; m >= 1; m >>= 1) {
        sum += __shfl_xor(sum, m, 32);
        sq  += __shfl_xor(sq,  m, 32);
    }
    float mean = sum * (1.f / DIMc);
    float var  = sq * (1.f / DIMc) - mean * mean;        // biased (matches jnp.var)
    float rstd = rsqrtf(var + LN_EPS);
#pragma unroll
    for (int i = 0; i < 8; ++i) {
#pragma unroll
        for (int j = 0; j < 4; ++j) {
            int c = lane * 4 + i * 128 + j;
            float o = (vals[i * 4 + j] - mean) * rstd * gam[c] + bet[c];
            xn[(size_t)row * DIMc + c] = to_bf16(o);
        }
    }
}

// ------------------- transpose + f32->bf16 weight convert ------------------
__global__ void tcvt_kernel(const float* __restrict__ src,
                            __bf16* __restrict__ dst, int K, int N) {
    long idx = (long)blockIdx.x * blockDim.x + threadIdx.x;
    if (idx >= (long)K * N) return;
    int n = (int)(idx / K);
    int k = (int)(idx - (long)n * K);
    dst[idx] = to_bf16(src[(size_t)k * N + n]);
}

// ------------------------------ QKV GEMM -----------------------------------
// C[4096 x 3072] = xn @ Wqkv ; per-wave 32(M) x 64(N) tile, K-step 32.
// Each B fragment feeds 2 WMMAs. Epilogue scatters to q/k/v_t (bf16).
__global__ __launch_bounds__(128) void qkv_gemm(const __bf16* __restrict__ xn,
                                                const __bf16* __restrict__ wT,
                                                __bf16* __restrict__ qb,
                                                __bf16* __restrict__ kb,
                                                __bf16* __restrict__ vtb) {
    int lane  = threadIdx.x;
    int mtile = blockIdx.y * blockDim.y + threadIdx.y;  // 0..127
    int c0    = blockIdx.x * 64;                        // 0..3008
    int row0  = mtile * 32;
    v8f zero = {0,0,0,0,0,0,0,0};
    v8f acc[2][4] = {{zero, zero, zero, zero}, {zero, zero, zero, zero}};
    for (int kk = 0; kk < DIMc / 32; ++kk) {
        int kpos = kk * 32;
        v16bf a0 = load_a16x32(xn + (size_t)(row0     ) * DIMc + kpos, DIMc);
        v16bf a1 = load_a16x32(xn + (size_t)(row0 + 16) * DIMc + kpos, DIMc);
#pragma unroll
        for (int t = 0; t < 4; ++t) {
            v16bf b = load_b32x16T(wT + (size_t)(c0 + 16 * t) * DIMc + kpos, DIMc);
            acc[0][t] = wmma_bf16(a0, b, acc[0][t]);
            acc[1][t] = wmma_bf16(a1, b, acc[1][t]);
        }
    }
    int g = lane >> 4, n = lane & 15;
#pragma unroll
    for (int qg = 0; qg < 2; ++qg) {
#pragma unroll
        for (int t = 0; t < 4; ++t) {
            int c   = c0 + 16 * t + n;
            int sel = c >> 10;          // 0=q 1=k 2=v
            int cc  = c & 1023;
            int h   = cc >> 6, d = cc & 63;
#pragma unroll
            for (int r = 0; r < 8; ++r) {
                int row = row0 + 16 * qg + r + 8 * g;
                int bb  = row >> 11, nq = row & 2047;
                size_t bh = (size_t)(bb * Hc + h);
                __bf16 val = to_bf16(acc[qg][t][r]);
                if (sel == 0)      qb [(bh * Nc + nq) * DHc + d] = val;
                else if (sel == 1) kb [(bh * Nc + nq) * DHc + d] = val;
                else               vtb[(bh * DHc + d) * Nc + nq] = val;
            }
        }
    }
}

// ------------------------------ Attention ----------------------------------
// Per wave: one 32-query tile of one (b,h). Stream keys in chunks of 32.
// Each K/V fragment feeds 2 WMMAs (two query sub-tiles).
__global__ __launch_bounds__(128) void attn_kernel(const __bf16* __restrict__ qb,
                                                   const __bf16* __restrict__ kb,
                                                   const __bf16* __restrict__ vtb,
                                                   __bf16* __restrict__ ao) {
    __shared__ __align__(16) __bf16 pbuf[4][32][32];    // 2 KB per wave
    int lane  = threadIdx.x;
    int wave  = blockIdx.x * blockDim.y + threadIdx.y;  // 0..2047
    int qtile = wave & 63;                              // 2048/32 query tiles
    int bh    = wave >> 6;                              // 0..31
    int b     = bh >> 4, h = bh & 15;
    const __bf16* Q  = qb  + (size_t)bh * Nc * DHc;
    const __bf16* K  = kb  + (size_t)bh * Nc * DHc;
    const __bf16* VT = vtb + (size_t)bh * DHc * Nc;
    int row0 = qtile * 32;

    v16bf aq[2][2];                                     // [qsub][dchunk]
#pragma unroll
    for (int qg = 0; qg < 2; ++qg) {
        aq[qg][0] = load_a16x32(Q + (size_t)(row0 + 16 * qg) * DHc + 0,  DHc);
        aq[qg][1] = load_a16x32(Q + (size_t)(row0 + 16 * qg) * DHc + 32, DHc);
    }

    v8f zero = {0,0,0,0,0,0,0,0};
    v8f o[2][4] = {{zero, zero, zero, zero}, {zero, zero, zero, zero}};
    v8f ds[2] = {zero, zero};
    __bf16 (*pb)[32] = pbuf[threadIdx.y];
    int g = lane >> 4, n = lane & 15;

    for (int kc = 0; kc < Nc / 32; ++kc) {
        int kbase = kc * 32;
        v16bf b00 = load_b32x16T(K + (size_t)(kbase     ) * DHc + 0,  DHc);
        v16bf b01 = load_b32x16T(K + (size_t)(kbase     ) * DHc + 32, DHc);
        v16bf b10 = load_b32x16T(K + (size_t)(kbase + 16) * DHc + 0,  DHc);
        v16bf b11 = load_b32x16T(K + (size_t)(kbase + 16) * DHc + 32, DHc);
#pragma unroll
        for (int qg = 0; qg < 2; ++qg) {
            v8f s0 = zero, s1 = zero;
            s0 = wmma_bf16(aq[qg][0], b00, s0);
            s0 = wmma_bf16(aq[qg][1], b01, s0);
            s1 = wmma_bf16(aq[qg][0], b10, s1);
            s1 = wmma_bf16(aq[qg][1], b11, s1);
#pragma unroll
            for (int r = 0; r < 8; ++r) {
                float e0 = __expf(SCALEc * s0[r]);
                float e1 = __expf(SCALEc * s1[r]);
                ds[qg][r] += e0 + e1;
                int m = 16 * qg + r + 8 * g;       // D-layout row
                pb[m][n]      = to_bf16(e0);       // keys kbase..kbase+15
                pb[m][16 + n] = to_bf16(e1);       // keys kbase+16..kbase+31
            }
        }
        __asm__ volatile("s_wait_dscnt 0" ::: "memory");
        // Re-read P in A-fragment layout (2 x 16x32 tiles).
        int ml = lane & 15, gl = lane >> 4;
        v16bf pa[2];
#pragma unroll
        for (int qg = 0; qg < 2; ++qg) {
            union { v16bf v; v8bf hh[2]; } up;
            up.hh[0] = *(const v8bf*)(&pb[16 * qg + ml][8 * gl]);
            up.hh[1] = *(const v8bf*)(&pb[16 * qg + ml][16 + 8 * gl]);
            pa[qg] = up.v;
        }
#pragma unroll
        for (int t = 0; t < 4; ++t) {
            v16bf bv = load_b32x16T(VT + (size_t)(16 * t) * Nc + kbase, Nc);
            o[0][t] = wmma_bf16(pa[0], bv, o[0][t]);
            o[1][t] = wmma_bf16(pa[1], bv, o[1][t]);
        }
    }
    // Row-sum reduction over the 16 N-lanes of each half-wave; lane-group
    // mapping of ds matches o (both D-layout) so the divide lines up.
#pragma unroll
    for (int qg = 0; qg < 2; ++qg) {
#pragma unroll
        for (int r = 0; r < 8; ++r) {
            float s = ds[qg][r];
            s += __shfl_xor(s, 1, 32);
            s += __shfl_xor(s, 2, 32);
            s += __shfl_xor(s, 4, 32);
            s += __shfl_xor(s, 8, 32);
            float inv = 1.f / (s + ATTN_EPS);
            o[qg][0][r] *= inv; o[qg][1][r] *= inv;
            o[qg][2][r] *= inv; o[qg][3][r] *= inv;
        }
    }
    // ao is [B*N][INNER] with col = h*64 + d  (b n (h d) layout).
#pragma unroll
    for (int qg = 0; qg < 2; ++qg) {
#pragma unroll
        for (int t = 0; t < 4; ++t) {
#pragma unroll
            for (int r = 0; r < 8; ++r) {
                int nq  = row0 + 16 * qg + r + 8 * g;
                int col = h * 64 + 16 * t + n;
                ao[(size_t)(b * Nc + nq) * INNERc + col] = to_bf16(o[qg][t][r]);
            }
        }
    }
}

// ------------------------------ Output GEMM --------------------------------
// out[4096 x 1024] = ao @ w_out + b_out   (f32 result), 32x64 wave tiles.
__global__ __launch_bounds__(128) void out_gemm(const __bf16* __restrict__ ao,
                                                const __bf16* __restrict__ wT,
                                                const float* __restrict__ bias,
                                                float* __restrict__ out) {
    int lane  = threadIdx.x;
    int mtile = blockIdx.y * blockDim.y + threadIdx.y;  // 0..127
    int c0    = blockIdx.x * 64;
    int row0  = mtile * 32;
    v8f zero = {0,0,0,0,0,0,0,0};
    v8f acc[2][4] = {{zero, zero, zero, zero}, {zero, zero, zero, zero}};
    for (int kk = 0; kk < INNERc / 32; ++kk) {
        int kpos = kk * 32;
        v16bf a0 = load_a16x32(ao + (size_t)(row0     ) * INNERc + kpos, INNERc);
        v16bf a1 = load_a16x32(ao + (size_t)(row0 + 16) * INNERc + kpos, INNERc);
#pragma unroll
        for (int t = 0; t < 4; ++t) {
            v16bf b = load_b32x16T(wT + (size_t)(c0 + 16 * t) * INNERc + kpos, INNERc);
            acc[0][t] = wmma_bf16(a0, b, acc[0][t]);
            acc[1][t] = wmma_bf16(a1, b, acc[1][t]);
        }
    }
    int g = lane >> 4, n = lane & 15;
#pragma unroll
    for (int qg = 0; qg < 2; ++qg) {
#pragma unroll
        for (int t = 0; t < 4; ++t) {
            int c = c0 + 16 * t + n;
            float bs = bias[c];
#pragma unroll
            for (int r = 0; r < 8; ++r) {
                int row = row0 + 16 * qg + r + 8 * g;
                out[(size_t)row * DIMc + c] = acc[qg][t][r] + bs;
            }
        }
    }
}

// ---------------------------------------------------------------------------
extern "C" void kernel_launch(void* const* d_in, const int* in_sizes, int n_in,
                              void* d_out, int out_size, void* d_ws, size_t ws_size,
                              hipStream_t stream) {
    (void)in_sizes; (void)n_in; (void)out_size; (void)ws_size;
    const float* x      = (const float*)d_in[0];
    const float* gamma  = (const float*)d_in[1];
    const float* beta   = (const float*)d_in[2];
    const float* w_qkv  = (const float*)d_in[3];
    const float* w_out  = (const float*)d_in[4];
    const float* b_out  = (const float*)d_in[5];
    float* out = (float*)d_out;

    char* ws = (char*)d_ws;
    size_t off = 0;
    auto grab = [&](size_t elems) { __bf16* p = (__bf16*)(ws + off);
                                    off += elems * sizeof(__bf16); return p; };
    const size_t ROWS = (size_t)Bc * Nc;                 // 4096
    __bf16* xn    = grab(ROWS * DIMc);                   // 8 MB
    __bf16* wqkvT = grab((size_t)3 * INNERc * DIMc);     // 6 MB  [3072][1024]
    __bf16* woutT = grab((size_t)DIMc * INNERc);         // 2 MB  [1024][1024]
    __bf16* qb    = grab((size_t)Bc * Hc * Nc * DHc);    // 8 MB
    __bf16* kb    = grab((size_t)Bc * Hc * Nc * DHc);    // 8 MB
    __bf16* vtb   = grab((size_t)Bc * Hc * DHc * Nc);    // 8 MB (transposed V)
    __bf16* ao    = grab(ROWS * INNERc);                 // 8 MB

    dim3 wblk(32, 4);   // 4 wave32s per block

    ln_kernel  <<<dim3((int)(ROWS / 4)), wblk, 0, stream>>>(x, gamma, beta, xn);
    tcvt_kernel<<<dim3((3 * INNERc * DIMc + 255) / 256), 256, 0, stream>>>(
        w_qkv, wqkvT, DIMc, 3 * INNERc);
    tcvt_kernel<<<dim3((DIMc * INNERc + 255) / 256), 256, 0, stream>>>(
        w_out, woutT, INNERc, DIMc);
    // 3072/64 = 48 column tiles, 4096/32/4 = 32 row-tile blocks
    qkv_gemm   <<<dim3(48, 32), wblk, 0, stream>>>(xn, wqkvT, qb, kb, vtb);
    // 2048 waves: 32 (b,h) x 64 query tiles of 32
    attn_kernel<<<dim3(512), wblk, 0, stream>>>(qb, kb, vtb, ao);
    // 1024/64 = 16 column tiles, 32 row-tile blocks
    out_gemm   <<<dim3(16, 32), wblk, 0, stream>>>(ao, woutT, b_out, out);
}
